// MACE_Ewald_7361573945909
// MI455X (gfx1250) — matrix operand order repaired and checked by reading, not compiled, not run on hardware
//
#include <hip/hip_runtime.h>
#include <math.h>

typedef __attribute__((ext_vector_type(16))) _Float16 v16h;
typedef __attribute__((ext_vector_type(8)))  float    v8f;

#define CCH 128
#define NLM 16
#define BG  8
#define PI_F 3.14159265358979323846f
#define SKIPF 0.5773502691896258f   /* (2+1)^-0.5 */

__device__ __forceinline__ float siluf(float x){ return x / (1.0f + __expf(-x)); }

__device__ __forceinline__ void atomAddF(float* p, float v){
#if defined(__HIP_DEVICE_COMPILE__) && __has_builtin(__hip_atomic_fetch_add)
  __hip_atomic_fetch_add(p, v, __ATOMIC_RELAXED, __HIP_MEMORY_SCOPE_AGENT);
#else
  atomicAdd(p, v);
#endif
}

/* ---------------- generic zero ---------------- */
__global__ void k_zero(float* __restrict__ p, size_t n){
  size_t i = (size_t)blockIdx.x * blockDim.x + threadIdx.x;
  if (i < n) p[i] = 0.0f;
}

/* ---------------- weight packing into WMMA B-fragment layout ----------------
 * B (K x N, row-major fp32) -> per (kt,nt) tiles of f16:
 *   packed[((kt*NT+nt)*32 + lane)*16 + e] = W[kt*32 + 16*(e>=8) + 8*(lane>=16) + (e&7)][nt*16 + (lane&15)]
 * matching the A-fragment k order (lanes 0-15 hold K=0..7,16..23, lanes 16-31 K=8..15,24..31).
 * K is zero-padded to a multiple of 32.
 */
__global__ void k_pack_b(const float* __restrict__ W, _Float16* __restrict__ Bp, int K, int Ncols){
  int idx = blockIdx.x * blockDim.x + threadIdx.x;
  int NT = Ncols >> 4;
  int KT = (K + 31) >> 5;
  int tot = KT * NT * 512;
  if (idx >= tot) return;
  int e    = idx & 15;
  int lane = (idx >> 4) & 31;
  int tile = idx >> 9;
  int nt = tile % NT;
  int kt = tile / NT;
  int n = (nt << 4) + (lane & 15);
  int k = (kt << 5) + ((e >> 3) << 4) + ((lane >> 4) << 3) + (e & 7);
  Bp[idx] = (k < K) ? (_Float16)W[(size_t)k * Ncols + n] : (_Float16)0.0f;
}

/* ---------------- WMMA GEMM ----------------
 * Cout = oscale * ( act(A@B + bias) + resid + resid2 )
 * A: M x K fp32, row stride lda, K multiple of 32, M multiple of 16,
 * Ncols multiple of 32 (so every wave owns exactly TWO adjacent 16x16 tiles:
 * one A fragment feeds two unconditional v_wmma_f32_16x16x32_f16 per K-step).
 */
__global__ void __launch_bounds__(128)
k_wmma_gemm(const float* __restrict__ A, const _Float16* __restrict__ Bp,
            const float* __restrict__ bias, const float* __restrict__ resid,
            const float* __restrict__ resid2, float* __restrict__ Cout,
            int M, int Ncols, int K, int lda, int act, float oscale){
  int wave = threadIdx.x >> 5;
  int lane = threadIdx.x & 31;
  int NT = Ncols >> 4;                   /* even by contract */
  int nt0 = (blockIdx.y * 4 + wave) * 2;
  if (nt0 >= NT) return;                 /* wave-uniform exit: EXEC stays all-ones */
  int mt = blockIdx.x;

  int row0 = (mt << 4) + (lane & 15);    /* A row for this lane (M = lane%16) */
  int koff = (lane >> 4) << 3;           /* +8 for upper half-wave (A layout)  */
  const float* arow = A + (size_t)row0 * lda;
  int KT = K >> 5;
  const _Float16* b0 = Bp + (size_t)nt0 * 512 + (size_t)lane * 16;

  v8f acc0 = {0.f,0.f,0.f,0.f,0.f,0.f,0.f,0.f};
  v8f acc1 = {0.f,0.f,0.f,0.f,0.f,0.f,0.f,0.f};
  for (int kt = 0; kt < KT; ++kt){
    int kb = (kt << 5) + koff;
    v16h a;
#pragma unroll
    for (int j = 0; j < 8; ++j) a[j]   = (_Float16)arow[kb + j];
#pragma unroll
    for (int j = 0; j < 8; ++j) a[8+j] = (_Float16)arow[kb + 16 + j];
    const _Float16* bk = b0 + (size_t)kt * NT * 512;
    v16h bf0 = *(const v16h*)(bk);
    v16h bf1 = *(const v16h*)(bk + 512);
    acc0 = __builtin_amdgcn_wmma_f32_16x16x32_f16(false, a, false, bf0, (short)0, acc0, false, false);
    acc1 = __builtin_amdgcn_wmma_f32_16x16x32_f16(false, a, false, bf1, (short)0, acc1, false, false);
  }

  /* C/D layout: rows v + 8*(lane>=16), col = lane%16 */
  int rb  = (mt << 4) + ((lane >> 4) << 3);
#pragma unroll
  for (int s = 0; s < 2; ++s){
    int col = ((nt0 + s) << 4) + (lane & 15);
    float bv = bias ? bias[col] : 0.0f;
    const v8f& acc = s ? acc1 : acc0;
    size_t o = (size_t)rb * (size_t)Ncols + (size_t)col;   /* running offset */
#pragma unroll
    for (int v = 0; v < 8; ++v){
      float x = acc[v] + bv;
      if (act) x = siluf(x);
      if (resid)  x += resid[o];
      if (resid2) x += resid2[o];
      Cout[o] = x * oscale;
      o += (size_t)Ncols;
    }
  }
}

/* ---------------- node embedding h = node_attrs @ W_embed ---------------- */
__global__ void k_embed(const float* __restrict__ attrs, const float* __restrict__ W,
                        float* __restrict__ h, int N){
  int t = blockIdx.x * blockDim.x + threadIdx.x;
  if (t >= N * CCH) return;
  int n = t / CCH, c = t % CCH;
  float acc = 0.f;
#pragma unroll
  for (int el = 0; el < 10; ++el) acc += attrs[n*10+el] * W[el*CCH + c];
  h[t] = acc;
}

/* ---------------- e0 = segment_sum(node_attrs @ atomic_E) ---------------- */
__global__ void k_e0(const float* __restrict__ attrs, const float* __restrict__ aE,
                     const int* __restrict__ batch, float* __restrict__ energy, int N){
  int n = blockIdx.x * blockDim.x + threadIdx.x;
  if (n >= N) return;
  float acc = 0.f;
#pragma unroll
  for (int el = 0; el < 10; ++el) acc += attrs[n*10+el] * aE[el];
  atomAddF(&energy[batch[n]], acc);
}

/* ---------------- edge geometry: Y (16 sph harm) + ef (8 radial, stride 32) ---- */
__global__ void k_edge_geom(const float* __restrict__ pos, const float* __restrict__ shifts,
                            const int* __restrict__ ei, float* __restrict__ Y,
                            float* __restrict__ ef, int E){
  int e = blockIdx.x * blockDim.x + threadIdx.x;
  if (e >= E) return;
  int s = ei[e], d = ei[E + e];
  float vx = pos[d*3+0] - pos[s*3+0] + shifts[e*3+0];
  float vy = pos[d*3+1] - pos[s*3+1] + shifts[e*3+1];
  float vz = pos[d*3+2] - pos[s*3+2] + shifts[e*3+2];
  float r = sqrtf(vx*vx + vy*vy + vz*vz);
  float inv = 1.0f / fmaxf(r, 1e-9f);
  float x = vx*inv, y = vy*inv, z = vz*inv;
  const float s3 = 1.7320508075688772f, s5 = 2.2360679774997896f, s15 = 3.872983346207417f;
  const float c70 = 2.091650066335189f, c105 = 10.246950765959598f;
  const float c42 = 1.620185174601965f, c7 = 1.3228756555322954f;
  float* yr = Y + (size_t)e * NLM;
  float zz = z*z, xx = x*x, yy = y*y;
  yr[0]  = 1.0f;
  yr[1]  = s3*x;  yr[2] = s3*y;  yr[3] = s3*z;
  yr[4]  = s15*x*y; yr[5] = s15*y*z; yr[6] = 0.5f*s5*(3.f*zz-1.f);
  yr[7]  = s15*x*z; yr[8] = 0.5f*s15*(xx-yy);
  yr[9]  = c70*y*(3.f*xx-yy); yr[10] = c105*x*y*z; yr[11] = c42*y*(5.f*zz-1.f);
  yr[12] = c7*z*(5.f*zz-3.f); yr[13] = c42*x*(5.f*zz-1.f);
  yr[14] = 0.5f*c105*z*(xx-yy); yr[15] = c70*x*(xx-3.f*yy);
  /* radial: sqrt(2/5)*sin(n*pi*r/5)/max(r,1e-9) * polynomial cutoff */
  float u = fminf(fmaxf(r * 0.2f, 0.f), 1.f);
  float u5 = u*u*u*u*u;
  float env = 1.f - 21.f*u5 + 35.f*u5*u - 15.f*u5*u*u;
  env *= (r < 5.0f) ? 1.0f : 0.0f;
  float pref = 0.6324555320336759f /* sqrt(0.4) */ * inv * env;
  float* er = ef + (size_t)e * 32;          /* zero-padded to K=32 for WMMA */
#pragma unroll
  for (int nb = 1; nb <= 8; ++nb) er[nb-1] = pref * __sinf((float)nb * (PI_F/5.0f) * r);
}

/* ---------------- sd[n] = prod_d sinc(0.1*pos[n,d]) (normalized sinc) ---------------- */
__global__ void k_sd(const float* __restrict__ pos, float* __restrict__ sd, int N){
  int n = blockIdx.x * blockDim.x + threadIdx.x;
  if (n >= N) return;
  float p = 1.0f;
#pragma unroll
  for (int d = 0; d < 3; ++d){
    float px = PI_F * 0.1f * pos[n*3+d];
    p *= (fabsf(px) < 1e-6f) ? 1.0f : __sinf(px)/px;
  }
  sd[n] = p;
}

/* ---------------- cosd/sind = sd * cos/sin(pos . k) ---------------- */
__global__ void k_phase(const float* __restrict__ pos, const float* __restrict__ kgrid,
                        const float* __restrict__ sd, float* __restrict__ cosd,
                        float* __restrict__ sind, int N, int NK){
  int t = blockIdx.x * blockDim.x + threadIdx.x;
  if (t >= N * NK) return;
  int n = t / NK, k = t % NK;
  float dp = pos[n*3+0]*kgrid[k*3+0] + pos[n*3+1]*kgrid[k*3+1] + pos[n*3+2]*kgrid[k*3+2];
  float sv = sd[n];
  cosd[t] = sv * __cosf(dp);
  sind[t] = sv * __sinf(dp);
}

/* ---------------- kdown = krbf @ Wdown (NK x 8) ---------------- */
__global__ void k_kdown(const float* __restrict__ krbf, const float* __restrict__ Wdown,
                        float* __restrict__ kdown, int NK){
  int t = blockIdx.x * blockDim.x + threadIdx.x;
  if (t >= NK * 8) return;
  int k = t / 8, p = t % 8;
  float acc = 0.f;
  for (int j = 0; j < 128; ++j) acc += krbf[k*128+j] * Wdown[j*8+p];
  kdown[t] = acc;
}

/* ---------------- kfilter = kdown @ WupE[layer] (NK x 128) ---------------- */
__global__ void k_kfilter(const float* __restrict__ kdown, const float* __restrict__ WupE_l,
                          float* __restrict__ kf, int NK){
  int t = blockIdx.x * blockDim.x + threadIdx.x;
  if (t >= NK * CCH) return;
  int k = t / CCH, c = t % CCH;
  float acc = 0.f;
#pragma unroll
  for (int p = 0; p < 8; ++p) acc += kdown[k*8+p] * WupE_l[p*CCH + c];
  kf[t] = acc;
}

/* ---------------- sfr/sfi: per-graph segment sums (batch sorted -> sequential sweep) */
__global__ void k_sf(const float* __restrict__ hres, const float* __restrict__ cosd,
                     const float* __restrict__ sind, const int* __restrict__ batch,
                     float* __restrict__ sfr, float* __restrict__ sfi, int N, int NK){
  int t = blockIdx.x * blockDim.x + threadIdx.x;
  if (t >= NK * CCH) return;
  int k = t / CCH, c = t % CCH;
  float accC = 0.f, accS = 0.f;
  int curg = batch[0];
  for (int n = 0; n < N; ++n){
    int g = batch[n];                         /* uniform across wave -> no divergence */
    if (g != curg){
      sfr[((size_t)curg*NK + k)*CCH + c] = accC;
      sfi[((size_t)curg*NK + k)*CCH + c] = accS;
      accC = 0.f; accS = 0.f; curg = g;
    }
    if ((n & 15) == 0 && n + 16 < N) __builtin_prefetch(&hres[(size_t)(n+16)*CCH + c], 0, 1);
    float hv = hres[(size_t)n*CCH + c];
    accC += cosd[(size_t)n*NK + k] * hv;
    accS += sind[(size_t)n*NK + k] * hv;
  }
  sfr[((size_t)curg*NK + k)*CCH + c] = accC;
  sfi[((size_t)curg*NK + k)*CCH + c] = accS;
}

/* ---------------- he = 0.01 * sum_k [ cosd*sfr*kf + sind*sfi*kf ] ---------------- */
__global__ void k_he(const float* __restrict__ cosd, const float* __restrict__ sind,
                     const float* __restrict__ sfr, const float* __restrict__ sfi,
                     const float* __restrict__ kf, const int* __restrict__ batch,
                     float* __restrict__ he, int N, int NK){
  int t = blockIdx.x * blockDim.x + threadIdx.x;
  if (t >= N * CCH) return;
  int n = t / CCH, c = t % CCH;
  int g = batch[n];
  float acc = 0.f;
  for (int k = 0; k < NK; ++k){
    float f = kf[k*CCH + c];
    acc += cosd[(size_t)n*NK + k] * sfr[((size_t)g*NK + k)*CCH + c] * f;
    acc += sind[(size_t)n*NK + k] * sfi[((size_t)g*NK + k)*CCH + c] * f;
  }
  he[t] = 0.01f * acc;
}

/* ---------------- message scatter: A[dst,c,lm] += w4[e,c*4+l]*hu[src,c]*Y[e,lm]/16 */
__global__ void k_msg(const float* __restrict__ w4, const float* __restrict__ hu,
                      const float* __restrict__ Y, const int* __restrict__ ei,
                      float* __restrict__ A, int E){
  size_t t = (size_t)blockIdx.x * blockDim.x + threadIdx.x;
  if (t >= (size_t)E * CCH) return;
  int e = (int)(t >> 7);
  int c = (int)(t & 127);
  int s = ei[e], d = ei[E + e];
  float hs = hu[(size_t)s*CCH + c] * 0.0625f;   /* fold 1/AVG_NEI */
  const float* wr = w4 + (size_t)e*512 + (size_t)c*4;
  const float* yr = Y + (size_t)e*NLM;
  float* ar = A + ((size_t)d*CCH + c)*NLM;
#pragma unroll
  for (int lm = 0; lm < NLM; ++lm){
    int l = (lm > 0) + (lm > 3) + (lm > 8);
    atomAddF(&ar[lm], wr[l] * hs * yr[lm]);
  }
}

/* ---------------- feats from A, w2, w3 ---------------- */
__global__ void k_feats(const float* __restrict__ A, const float* __restrict__ w2l,
                        const float* __restrict__ w3l, float* __restrict__ feats, int N){
  int t = blockIdx.x * blockDim.x + threadIdx.x;
  if (t >= N * CCH) return;
  int n = t / CCH, c = t % CCH;
  const float* a = A + ((size_t)n*CCH + c)*NLM;
  float i0 = a[0]*a[0];
  float i1 = a[1]*a[1] + a[2]*a[2] + a[3]*a[3];
  float i2 = 0.f, i3 = 0.f;
#pragma unroll
  for (int s = 4; s <= 8; ++s)  i2 += a[s]*a[s];
#pragma unroll
  for (int s = 9; s <= 15; ++s) i3 += a[s]*a[s];
  float t2 = i0*w2l[c] + i1*w2l[CCH+c] + i2*w2l[2*CCH+c] + i3*w2l[3*CCH+c];
  float t3 = i0*w3l[c] + i1*w3l[CCH+c] + i2*w3l[2*CCH+c] + i3*w3l[3*CCH+c];
  float scal = a[0];
  feats[t] = scal + t2 + scal*t3;
}

/* ---------------- readouts ---------------- */
__global__ void k_readout0(const float* __restrict__ h, const float* __restrict__ Wr0,
                           const int* __restrict__ batch, float* __restrict__ energy, int N){
  int n = blockIdx.x * blockDim.x + threadIdx.x;
  if (n >= N) return;
  float acc = 0.f;
  for (int c = 0; c < CCH; ++c) acc += h[(size_t)n*CCH + c] * Wr0[c];
  atomAddF(&energy[batch[n]], acc);
}

__global__ void k_readout1(const float* __restrict__ h, const float* __restrict__ Wr1a,
                           const float* __restrict__ Wr1b, const int* __restrict__ batch,
                           float* __restrict__ energy, int N){
  int n = blockIdx.x * blockDim.x + threadIdx.x;
  if (n >= N) return;
  float tj[16];
#pragma unroll
  for (int j = 0; j < 16; ++j) tj[j] = 0.f;
  for (int c = 0; c < CCH; ++c){
    float hv = h[(size_t)n*CCH + c];
#pragma unroll
    for (int j = 0; j < 16; ++j) tj[j] += hv * Wr1a[c*16 + j];
  }
  float acc = 0.f;
#pragma unroll
  for (int j = 0; j < 16; ++j) acc += siluf(tj[j]) * Wr1b[j];
  atomAddF(&energy[batch[n]], acc);
}

/* =========================================================================== */
extern "C" void kernel_launch(void* const* d_in, const int* in_sizes, int n_in,
                              void* d_out, int out_size, void* d_ws, size_t ws_size,
                              hipStream_t stream) {
  const float* positions = (const float*)d_in[0];
  const float* node_attrs= (const float*)d_in[1];
  const float* shifts    = (const float*)d_in[2];
  const int*   edge_index= (const int*)d_in[3];
  const int*   batch     = (const int*)d_in[4];
  const float* kgrid     = (const float*)d_in[5];
  const float* krbf      = (const float*)d_in[6];
  const float* W_embed   = (const float*)d_in[7];
  const float* atomic_E  = (const float*)d_in[8];
  const float* rW1  = (const float*)d_in[9];
  const float* rb1  = (const float*)d_in[10];
  const float* rW2  = (const float*)d_in[11];
  const float* rb2  = (const float*)d_in[12];
  const float* rW3  = (const float*)d_in[13];
  const float* rb3  = (const float*)d_in[14];
  const float* rW4  = (const float*)d_in[15];
  const float* Wup  = (const float*)d_in[16];
  const float* w2w  = (const float*)d_in[17];
  const float* w3w  = (const float*)d_in[18];
  const float* Wmix = (const float*)d_in[19];
  const float* Wr0  = (const float*)d_in[20];
  const float* Wr1a = (const float*)d_in[21];
  const float* Wr1b = (const float*)d_in[22];
  const float* Wdown= (const float*)d_in[23];
  const float* WupE = (const float*)d_in[24];
  const float* Wpre1= (const float*)d_in[25];
  const float* bpre1= (const float*)d_in[26];
  const float* Wpre2= (const float*)d_in[27];
  const float* bpre2= (const float*)d_in[28];
  const float* Wm1  = (const float*)d_in[29];
  const float* bm1  = (const float*)d_in[30];
  const float* Wm2  = (const float*)d_in[31];
  const float* bm2  = (const float*)d_in[32];

  const int N  = in_sizes[0] / 3;     /* 3200 */
  const int E  = in_sizes[3] / 2;     /* 51200 */
  const int NK = in_sizes[5] / 3;     /* ~123 k-points */
  float* energy = (float*)d_out;

  /* ---- workspace carve-up (entire working set fits 192MB L2) ---- */
  char* base = (char*)d_ws;
  size_t off = 0;
  auto falloc = [&](size_t elems)->float*{
    float* p = (float*)(base + off); off += ((elems*sizeof(float) + 255) & ~(size_t)255); return p; };
  auto halloc = [&](size_t elems)->_Float16*{
    _Float16* p = (_Float16*)(base + off); off += ((elems*sizeof(_Float16) + 255) & ~(size_t)255); return p; };

  float* Yb    = falloc((size_t)E * NLM);
  float* efb   = falloc((size_t)E * 32);     /* zero-padded K=32 */
  float* hb    = falloc((size_t)N * CCH);
  float* t1b   = falloc((size_t)N * CCH);
  float* hresb = falloc((size_t)N * CCH);
  float* heA   = falloc((size_t)N * CCH);
  float* heB   = falloc((size_t)N * CCH);
  float* hub   = falloc((size_t)N * CCH);
  float* featsb= falloc((size_t)N * CCH);
  float* sdb   = falloc((size_t)N);
  float* cosdb = falloc((size_t)N * NK);
  float* sindb = falloc((size_t)N * NK);
  float* sfrb  = falloc((size_t)BG * NK * CCH);
  float* sfib  = falloc((size_t)BG * NK * CCH);
  float* kdownb= falloc((size_t)NK * 8);
  float* kfb   = falloc((size_t)2 * NK * CCH);
  float* e1b   = falloc((size_t)E * 64);
  float* e2b   = falloc((size_t)E * 64);
  float* w4b   = falloc((size_t)E * 512);
  float* Ab    = falloc((size_t)N * CCH * NLM);

  const size_t s128 = 4*8*512;   /* packed halves for 128x128 */
  const size_t sR1  = 1*4*512;   /* 8(->32)x64 */
  const size_t sR2  = 2*4*512;   /* 64x64 */
  const size_t sR4  = 2*32*512;  /* 64x512 */
  _Float16 *pWpre1[2], *pWpre2[2], *pWm1[2], *pWm2[2], *pWupL[2], *pWmixL[2];
  _Float16 *pR1[2], *pR2[2], *pR3[2], *pR4[2];
  for (int i = 0; i < 2; ++i){
    pWpre1[i] = halloc(s128); pWpre2[i] = halloc(s128);
    pWm1[i]   = halloc(s128); pWm2[i]   = halloc(s128);
    pWupL[i]  = halloc(s128); pWmixL[i] = halloc(s128);
    pR1[i] = halloc(sR1); pR2[i] = halloc(sR2); pR3[i] = halloc(sR2); pR4[i] = halloc(sR4);
  }

  auto blocks = [](size_t total){ return dim3((unsigned)((total + 255) / 256)); };
  auto pack = [&](const float* W, _Float16* dst, int K, int Ncols){
    int tot = ((K + 31)/32) * (Ncols/16) * 512;
    k_pack_b<<<blocks(tot), 256, 0, stream>>>(W, dst, K, Ncols);
  };
  /* grid: one wave per two adjacent n-tiles */
  auto gemm = [&](const float* A, const _Float16* Bp, const float* bias,
                  const float* resid, const float* resid2, float* Co,
                  int M, int Ncols, int K, int lda, int act, float oscale){
    int NT = Ncols / 16;
    dim3 g(M/16, (NT + 7)/8);
    k_wmma_gemm<<<g, 128, 0, stream>>>(A, Bp, bias, resid, resid2, Co,
                                       M, Ncols, K, lda, act, oscale);
  };

  /* ---- pack all weights (tiny) ---- */
  for (int i = 0; i < 2; ++i){
    pack(Wpre1 + (size_t)i*CCH*CCH, pWpre1[i], CCH, CCH);
    pack(Wpre2 + (size_t)i*CCH*CCH, pWpre2[i], CCH, CCH);
    pack(Wm1   + (size_t)i*CCH*CCH, pWm1[i],   CCH, CCH);
    pack(Wm2   + (size_t)i*CCH*CCH, pWm2[i],   CCH, CCH);
    pack(Wup   + (size_t)i*CCH*CCH, pWupL[i],  CCH, CCH);
    pack(Wmix  + (size_t)i*CCH*CCH, pWmixL[i], CCH, CCH);
    pack(rW1 + (size_t)i*8*64,   pR1[i], 8, 64);   /* zero-pads K 8..31 */
    pack(rW2 + (size_t)i*64*64,  pR2[i], 64, 64);
    pack(rW3 + (size_t)i*64*64,  pR3[i], 64, 64);
    pack(rW4 + (size_t)i*64*512, pR4[i], 64, 512);
  }

  /* ---- precompute ---- */
  k_zero<<<1, 256, 0, stream>>>(energy, (size_t)out_size);
  k_zero<<<blocks((size_t)E*32), 256, 0, stream>>>(efb, (size_t)E*32);
  k_e0<<<blocks(N), 256, 0, stream>>>(node_attrs, atomic_E, batch, energy, N);
  k_embed<<<blocks((size_t)N*CCH), 256, 0, stream>>>(node_attrs, W_embed, hb, N);
  k_edge_geom<<<blocks(E), 256, 0, stream>>>(positions, shifts, edge_index, Yb, efb, E);
  k_sd<<<blocks(N), 256, 0, stream>>>(positions, sdb, N);
  k_phase<<<blocks((size_t)N*NK), 256, 0, stream>>>(positions, kgrid, sdb, cosdb, sindb, N, NK);
  k_kdown<<<blocks((size_t)NK*8), 256, 0, stream>>>(krbf, Wdown, kdownb, NK);
  for (int i = 0; i < 2; ++i)
    k_kfilter<<<blocks((size_t)NK*CCH), 256, 0, stream>>>(kdownb, WupE + (size_t)i*8*CCH,
                                                          kfb + (size_t)i*NK*CCH, NK);

  /* ---- interaction layers ---- */
  for (int i = 0; i < 2; ++i){
    /* hres = h + silu(h@Wpre1+b)@Wpre2 + b */
    gemm(hb,  pWpre1[i], bpre1 + (size_t)i*CCH, nullptr, nullptr, t1b,   N, CCH, CCH, CCH, 1, 1.f);
    gemm(t1b, pWpre2[i], bpre2 + (size_t)i*CCH, hb,      nullptr, hresb, N, CCH, CCH, CCH, 0, 1.f);

    /* Ewald block */
    k_zero<<<blocks((size_t)BG*NK*CCH), 256, 0, stream>>>(sfrb, (size_t)BG*NK*CCH);
    k_zero<<<blocks((size_t)BG*NK*CCH), 256, 0, stream>>>(sfib, (size_t)BG*NK*CCH);
    k_sf<<<blocks((size_t)NK*CCH), 256, 0, stream>>>(hresb, cosdb, sindb, batch, sfrb, sfib, N, NK);
    k_he<<<blocks((size_t)N*CCH), 256, 0, stream>>>(cosdb, sindb, sfrb, sfib,
                                                    kfb + (size_t)i*NK*CCH, batch, heA, N, NK);
    gemm(heA, pWm1[i], bm1 + (size_t)i*CCH, nullptr, nullptr, heB, N, CCH, CCH, CCH, 1, 1.f);
    gemm(heB, pWm2[i], bm2 + (size_t)i*CCH, nullptr, nullptr, heA, N, CCH, CCH, CCH, 1, 1.f);

    /* message passing */
    gemm(hb,  pWupL[i], nullptr, nullptr, nullptr, hub, N, CCH, CCH, CCH, 0, 1.f);
    gemm(efb, pR1[i], rb1 + (size_t)i*64, nullptr, nullptr, e1b, E, 64, 32, 32, 1, 1.f);
    gemm(e1b, pR2[i], rb2 + (size_t)i*64, nullptr, nullptr, e2b, E, 64, 64, 64, 1, 1.f);
    gemm(e2b, pR3[i], rb3 + (size_t)i*64, nullptr, nullptr, e1b, E, 64, 64, 64, 1, 1.f);
    gemm(e1b, pR4[i], nullptr, nullptr, nullptr, w4b, E, 512, 64, 64, 0, 1.f);
    k_zero<<<blocks((size_t)N*CCH*NLM), 256, 0, stream>>>(Ab, (size_t)N*CCH*NLM);
    k_msg<<<blocks((size_t)E*CCH), 256, 0, stream>>>(w4b, hub, Yb, edge_index, Ab, E);
    k_feats<<<blocks((size_t)N*CCH), 256, 0, stream>>>(Ab, w2w + (size_t)i*4*CCH,
                                                       w3w + (size_t)i*4*CCH, featsb, N);
    /* h = SKIP * (feats@Wmix + h + he)  -- fused epilogue, writes h in place */
    gemm(featsb, pWmixL[i], nullptr, hb, heA, hb, N, CCH, CCH, CCH, 0, SKIPF);

    if (i == 0) k_readout0<<<blocks(N), 256, 0, stream>>>(hb, Wr0, batch, energy, N);
    else        k_readout1<<<blocks(N), 256, 0, stream>>>(hb, Wr1a, Wr1b, batch, energy, N);
  }
  (void)n_in; (void)ws_size;
}
